// UNICONDRANDLayer_68453188764130
// MI455X (gfx1250) — compile-verified, matrix operand order checked by reading, not run on hardware
//
#include <hip/hip_runtime.h>
#include <math.h>

// ---------------- problem constants ----------------
#define B_   128
#define S_   256
#define D_   256
#define H_   8
#define DH_  32
#define N_   (B_ * S_)         // 32768
#define E_   524288
#define ND_  ((size_t)N_ * D_) // 8388608
#define SCALE_ 0.17677669529663689f  // 1/sqrt(32)

// ---------------- WMMA types ----------------
typedef __attribute__((ext_vector_type(16))) __bf16         v16bf;
typedef __attribute__((ext_vector_type(16))) unsigned short v16u;
typedef __attribute__((ext_vector_type(8)))  float          v8f;

__device__ __forceinline__ unsigned short f2bf(float f) {
  unsigned u = __float_as_uint(f);
  u += 0x7FFFu + ((u >> 16) & 1u);   // round-to-nearest-even
  return (unsigned short)(u >> 16);
}

// K index held by fragment element e for a lane in half h (ISA 7.12.2, 16-bit A 16x32)
__device__ __forceinline__ int frag_k(int e, int h) {
  return ((e >> 3) << 4) + (h << 3) + (e & 7);
}

#define WMMA_BF16(a, b, c) \
  __builtin_amdgcn_wmma_f32_16x16x32_bf16(false, (a), false, (b), (short)0, (c), false, false)

// A fragment (16 rows x 32 K) from fp32 row-major, leading dim lda, scaled
__device__ __forceinline__ v16bf load_a_f32(const float* A, int lda, int row0, int k0,
                                            float scale, int lane) {
  int m = lane & 15, h = lane >> 4;
  const float* p = A + (size_t)(row0 + m) * lda + k0;
  v16u u;
#pragma unroll
  for (int e = 0; e < 16; ++e) u[e] = f2bf(p[frag_k(e, h)] * scale);
  return __builtin_bit_cast(v16bf, u);
}

// B fragment (32 K x 16 cols) from bf16(u16) row-major [K x Nout]
__device__ __forceinline__ v16bf load_b_u16(const unsigned short* W, int ldb, int k0,
                                            int col0, int lane) {
  int n = lane & 15, h = lane >> 4;
  v16u u;
#pragma unroll
  for (int e = 0; e < 16; ++e)
    u[e] = W[(size_t)(k0 + frag_k(e, h)) * ldb + col0 + n];
  return __builtin_bit_cast(v16bf, u);
}

// B fragment from fp32 row-major (B[k][n] = S[(k0+k)*ld + col0+n])
__device__ __forceinline__ v16bf load_b_f32(const float* S, int ld, int k0, int col0, int lane) {
  int n = lane & 15, h = lane >> 4;
  v16u u;
#pragma unroll
  for (int e = 0; e < 16; ++e)
    u[e] = f2bf(S[(size_t)(k0 + frag_k(e, h)) * ld + col0 + n]);
  return __builtin_bit_cast(v16bf, u);
}

// B fragment, transposed source: B[k][n] = S[(n0+n)*ld + k0+k]   (for Q @ K^T)
__device__ __forceinline__ v16bf load_bT_f32(const float* S, int ld, int n0, int k0, int lane) {
  int n = lane & 15, h = lane >> 4;
  const float* p = S + (size_t)(n0 + n) * ld + k0;
  v16u u;
#pragma unroll
  for (int e = 0; e < 16; ++e) u[e] = f2bf(p[frag_k(e, h)]);
  return __builtin_bit_cast(v16bf, u);
}

// ---------------- generic bf16-WMMA GEMM:  C = act(A @ W + bias) ----------------
// A: [M x K] fp32, W: [K x Nout] bf16(u16), C: [M x Nout] fp32.
// grid.x = M/128 (8 waves, each one 16-row tile), grid.y = Nout/16, block = 256.
__global__ void k_gemm_bf16(const float* __restrict__ A, const unsigned short* __restrict__ W,
                            const float* __restrict__ bias, float* __restrict__ C,
                            int M, int K, int Nout, int relu) {
  int lane = threadIdx.x & 31;
  int wave = threadIdx.x >> 5;
  int row0 = (blockIdx.x * 8 + wave) * 16;
  int col0 = blockIdx.y * 16;
  if (row0 >= M) return;

  v8f acc = {};
  for (int k0 = 0; k0 < K; k0 += 32) {
    if (k0 + 32 < K)
      __builtin_prefetch(A + (size_t)(row0 + (lane & 15)) * K + k0 + 32, 0, 1);
    v16bf a = load_a_f32(A, K, row0, k0, 1.0f, lane);
    v16bf b = load_b_u16(W, Nout, k0, col0, lane);
    acc = WMMA_BF16(a, b, acc);
  }

  int n = lane & 15, hh = lane >> 4;
  float bv = bias ? bias[col0 + n] : 0.0f;
#pragma unroll
  for (int r = 0; r < 8; ++r) {
    int m = r + hh * 8;
    float v = acc[r] + bv;
    if (relu) v = fmaxf(v, 0.0f);
    C[(size_t)(row0 + m) * Nout + col0 + n] = v;
  }
}

// ---------------- flash attention (one wave per (b, head, 16-query tile)) ----------------
// Q,K,V,O: [N_, D_] fp32, head h occupies columns h*32 .. h*32+31.
__global__ void k_flash_attn(const float* __restrict__ Q, const float* __restrict__ K,
                             const float* __restrict__ V, float* __restrict__ O) {
  __shared__ unsigned short lds_p[8][16 * 32];  // per-wave P tile (bf16)
  int lane = threadIdx.x & 31;
  int wave = threadIdx.x >> 5;
  int job  = blockIdx.x * 8 + wave;             // 16384 jobs total
  int qt   = job & 15;                          // query tile (S/16 = 16)
  int bh   = job >> 4;
  int head = bh & (H_ - 1);
  int b    = bh >> 3;

  const float* Qp = Q + ((size_t)b * S_ + qt * 16) * D_ + head * DH_;
  v16bf qfrag = load_a_f32(Qp, D_, 0, 0, SCALE_, lane);

  float mi[8], li[8];
#pragma unroll
  for (int r = 0; r < 8; ++r) { mi[r] = -3.0e38f; li[r] = 0.0f; }
  v8f o1 = {}, o2 = {};

  int n = lane & 15, hh = lane >> 4;
  unsigned short* lp = lds_p[wave];

  for (int kb = 0; kb < S_; kb += 32) {
    const float* Kp = K + ((size_t)b * S_ + kb) * D_ + head * DH_;
    const float* Vp = V + ((size_t)b * S_ + kb) * D_ + head * DH_;

    v16bf kb1 = load_bT_f32(Kp, D_, 0, 0, lane);   // keys kb..kb+15
    v16bf kb2 = load_bT_f32(Kp, D_, 16, 0, lane);  // keys kb+16..kb+31
    v8f s1 = {}, s2 = {};
    s1 = WMMA_BF16(qfrag, kb1, s1);
    s2 = WMMA_BF16(qfrag, kb2, s2);

#pragma unroll
    for (int r = 0; r < 8; ++r) {
      // row max over 32 keys (16 lanes within this half x 2 fragments)
      float t = fmaxf(s1[r], s2[r]);
      t = fmaxf(t, __shfl_xor(t, 8, 32));
      t = fmaxf(t, __shfl_xor(t, 4, 32));
      t = fmaxf(t, __shfl_xor(t, 2, 32));
      t = fmaxf(t, __shfl_xor(t, 1, 32));
      float mn = fmaxf(mi[r], t);
      float alpha = __expf(mi[r] - mn);
      mi[r] = mn;
      float p1 = __expf(s1[r] - mn);
      float p2 = __expf(s2[r] - mn);
      float rs = p1 + p2;
      rs += __shfl_xor(rs, 8, 32);
      rs += __shfl_xor(rs, 4, 32);
      rs += __shfl_xor(rs, 2, 32);
      rs += __shfl_xor(rs, 1, 32);
      li[r] = li[r] * alpha + rs;
      o1[r] *= alpha;
      o2[r] *= alpha;
      s1[r] = p1;
      s2[r] = p2;
    }

    // C-layout -> A-layout transpose of P through LDS (bf16)
#pragma unroll
    for (int r = 0; r < 8; ++r) {
      int m = r + hh * 8;
      lp[m * 32 + n]      = f2bf(s1[r]);
      lp[m * 32 + n + 16] = f2bf(s2[r]);
    }
    asm volatile("s_wait_dscnt 0" ::: "memory");
    {
      int m = lane & 15, h = lane >> 4;
      v16u pu;
#pragma unroll
      for (int e = 0; e < 16; ++e) pu[e] = lp[m * 32 + frag_k(e, h)];
      v16bf pf = __builtin_bit_cast(v16bf, pu);
      v16bf vb1 = load_b_f32(Vp, D_, 0, 0, lane);   // dh 0..15
      v16bf vb2 = load_b_f32(Vp, D_, 0, 16, lane);  // dh 16..31
      o1 = WMMA_BF16(pf, vb1, o1);
      o2 = WMMA_BF16(pf, vb2, o2);
    }
    asm volatile("s_wait_dscnt 0" ::: "memory");
  }

  float* Op = O + ((size_t)b * S_ + qt * 16) * D_ + head * DH_;
#pragma unroll
  for (int r = 0; r < 8; ++r) {
    int m = r + hh * 8;
    float inv = 1.0f / li[r];
    Op[(size_t)m * D_ + n]      = o1[r] * inv;
    Op[(size_t)m * D_ + n + 16] = o2[r] * inv;
  }
}

// ---------------- small / elementwise kernels ----------------
__global__ void k_cvt_bf16(const float* __restrict__ in, unsigned short* __restrict__ out, size_t n) {
  size_t i = (size_t)blockIdx.x * blockDim.x + threadIdx.x;
  if (i < n) out[i] = f2bf(in[i]);
}

__global__ void k_deg_init(float* deg) {
  size_t i = (size_t)blockIdx.x * blockDim.x + threadIdx.x;
  if (i < N_) deg[i] = 1.0f;  // self loop
}
__global__ void k_deg_edges(const long long* __restrict__ col, float* deg) {
  size_t e = (size_t)blockIdx.x * blockDim.x + threadIdx.x;
  if (e < E_) atomicAdd(&deg[col[e]], 1.0f);
}
__global__ void k_dinv(float* deg) {
  size_t i = (size_t)blockIdx.x * blockDim.x + threadIdx.x;
  if (i < N_) deg[i] = rsqrtf(deg[i]);
}

// h_gcn = dinv^2 * xw (self-loop term), agg = 0
__global__ void k_gcn_init(const float* __restrict__ xw, const float* __restrict__ dinv,
                           float* __restrict__ hg, float* __restrict__ agg) {
  size_t i = (size_t)blockIdx.x * blockDim.x + threadIdx.x;
  if (i >= ND_) return;
  float di = dinv[i >> 8];  // D_=256
  hg[i] = di * di * xw[i];
  agg[i] = 0.0f;
}

// fused scatter: GCN (norm * xw[row] -> hg[col]) and GIN (x[row] -> agg[col])
__global__ void k_scatter(const long long* __restrict__ row, const long long* __restrict__ col,
                          const float* __restrict__ dinv, const float* __restrict__ x,
                          const float* __restrict__ xw, float* __restrict__ hg,
                          float* __restrict__ agg) {
  size_t idx = (size_t)blockIdx.x * blockDim.x + threadIdx.x;
  if (idx >= (size_t)E_ * (D_ / 4)) return;
  size_t e = idx >> 6;           // D/4 = 64 quads per edge
  int d = (int)(idx & 63) << 2;
  long long r = row[e], c = col[e];
  float w = dinv[r] * dinv[c];
  const float* xs  = x  + (size_t)r * D_ + d;
  const float* xws = xw + (size_t)r * D_ + d;
  float* hp = hg  + (size_t)c * D_ + d;
  float* ap = agg + (size_t)c * D_ + d;
#pragma unroll
  for (int t = 0; t < 4; ++t) {
    atomicAdd(&hp[t], w * xws[t]);
    atomicAdd(&ap[t], xs[t]);
  }
}

// per-column BatchNorm stats (biased var), one block per column
__global__ void k_bn_stats(const float* __restrict__ X, float* __restrict__ mean,
                           float* __restrict__ rstd, int nrows) {
  __shared__ float ss[256], sq[256];
  int j = blockIdx.x;
  float s = 0.0f, q = 0.0f;
  for (int i = threadIdx.x; i < nrows; i += blockDim.x) {
    float v = X[(size_t)i * D_ + j];
    s += v; q += v * v;
  }
  ss[threadIdx.x] = s; sq[threadIdx.x] = q;
  __syncthreads();
  for (int st = 128; st; st >>= 1) {
    if ((int)threadIdx.x < st) { ss[threadIdx.x] += ss[threadIdx.x + st]; sq[threadIdx.x] += sq[threadIdx.x + st]; }
    __syncthreads();
  }
  if (threadIdx.x == 0) {
    float m = ss[0] / nrows;
    float var = sq[0] / nrows - m * m;
    mean[j] = m;
    rstd[j] = rsqrtf(var + 1e-5f);
  }
}

__global__ void k_bn_apply(const float* __restrict__ in, float* __restrict__ out,
                           const float* __restrict__ mean, const float* __restrict__ rstd,
                           const float* __restrict__ g, const float* __restrict__ b, size_t n) {
  size_t i = (size_t)blockIdx.x * blockDim.x + threadIdx.x;
  if (i >= n) return;
  int j = (int)(i & (D_ - 1));
  out[i] = (in[i] - mean[j]) * rstd[j] * g[j] + b[j];
}

__global__ void k_add(const float* __restrict__ a, const float* __restrict__ b,
                      float* __restrict__ out, size_t n) {
  size_t i = (size_t)blockIdx.x * blockDim.x + threadIdx.x;
  if (i < n) out[i] = a[i] + b[i];
}

__global__ void k_gzero(float* g) {
  if (threadIdx.x == 0 && blockIdx.x == 0) { g[0] = 0.0f; g[1] = 0.0f; }
}
__global__ void k_reduce(const float* __restrict__ X, size_t n, float* g) {
  __shared__ float ss[256], sq[256];
  float s = 0.0f, q = 0.0f;
  for (size_t i = (size_t)blockIdx.x * blockDim.x + threadIdx.x; i < n;
       i += (size_t)gridDim.x * blockDim.x) {
    float v = X[i]; s += v; q += v * v;
  }
  ss[threadIdx.x] = s; sq[threadIdx.x] = q;
  __syncthreads();
  for (int st = 128; st; st >>= 1) {
    if ((int)threadIdx.x < st) { ss[threadIdx.x] += ss[threadIdx.x + st]; sq[threadIdx.x] += sq[threadIdx.x + st]; }
    __syncthreads();
  }
  if (threadIdx.x == 0) { atomicAdd(&g[0], ss[0]); atomicAdd(&g[1], sq[0]); }
}
__global__ void k_gfinal(float* g, float count) {
  if (threadIdx.x == 0 && blockIdx.x == 0) {
    float gm = g[0] / count;
    float var = (g[1] - count * gm * gm) / (count - 1.0f);  // unbiased (torch .std)
    float gs = sqrtf(fmaxf(var, 0.0f));
    g[2] = gm;
    g[3] = 1.0f / (gs + 1e-8f);
  }
}

// kv = 0.5*(e0-gm)/(gs+eps) + 0.5*(e1-gm)/(gs+eps) + 0.1*x
__global__ void k_kv(const float* __restrict__ e0, const float* __restrict__ e1,
                     const float* __restrict__ x, const float* __restrict__ g,
                     float* __restrict__ kv, size_t n) {
  size_t i = (size_t)blockIdx.x * blockDim.x + threadIdx.x;
  if (i >= n) return;
  float gm = g[2], inv = g[3];
  kv[i] = 0.5f * ((e0[i] - gm) + (e1[i] - gm)) * inv + 0.1f * x[i];
}

// ---------------- host-side orchestration ----------------
extern "C" void kernel_launch(void* const* d_in, const int* in_sizes, int n_in,
                              void* d_out, int out_size, void* d_ws, size_t ws_size,
                              hipStream_t stream) {
  (void)in_sizes; (void)n_in; (void)out_size; (void)ws_size;

  const float*     x     = (const float*)d_in[0];
  const long long* ei    = (const long long*)d_in[1];
  const long long* row   = ei;
  const long long* col   = ei + E_;
  const float* W_gcn = (const float*)d_in[2];
  const float* W_gin1 = (const float*)d_in[4];  const float* b_gin1 = (const float*)d_in[5];
  const float* W_gin2 = (const float*)d_in[6];  const float* b_gin2 = (const float*)d_in[7];
  const float* g_kv  = (const float*)d_in[8];   const float* be_kv  = (const float*)d_in[9];
  const float* Wq1   = (const float*)d_in[10];  const float* bq1    = (const float*)d_in[11];
  const float* Wq2   = (const float*)d_in[12];  const float* bq2    = (const float*)d_in[13];
  const float* Wq_a  = (const float*)d_in[14];  const float* bq_a   = (const float*)d_in[15];
  const float* Wk_a  = (const float*)d_in[16];  const float* bk_a   = (const float*)d_in[17];
  const float* Wv_a  = (const float*)d_in[18];  const float* bv_a   = (const float*)d_in[19];
  const float* Wo_a  = (const float*)d_in[20];  const float* bo_a   = (const float*)d_in[21];
  const float* g_at  = (const float*)d_in[22];  const float* be_at  = (const float*)d_in[23];
  const float* Wf1   = (const float*)d_in[24];  const float* bf1    = (const float*)d_in[25];
  const float* Wf2   = (const float*)d_in[26];  const float* bf2    = (const float*)d_in[27];
  const float* g2    = (const float*)d_in[28];  const float* be2    = (const float*)d_in[29];

  char* ws = (char*)d_ws;
  size_t off = 0;
  auto alloc = [&](size_t bytes) -> void* {
    void* p = ws + off;
    off += (bytes + 255) & ~(size_t)255;
    return p;
  };

  float* B1 = (float*)alloc(ND_ * 4);
  float* B2 = (float*)alloc(ND_ * 4);
  float* B3 = (float*)alloc(ND_ * 4);
  float* B4 = (float*)alloc(ND_ * 4);
  float* B5 = (float*)alloc(ND_ * 2 * 4);      // N x 2D (FFN hidden)
  float* deg    = (float*)alloc(N_ * 4);
  float* mean   = (float*)alloc(D_ * 4);
  float* rstd   = (float*)alloc(D_ * 4);
  float* gstats = (float*)alloc(8 * 4);

  const size_t DD = (size_t)D_ * D_;
  unsigned short* wb_gcn  = (unsigned short*)alloc(DD * 2);
  unsigned short* wb_gin1 = (unsigned short*)alloc(DD * 2);
  unsigned short* wb_gin2 = (unsigned short*)alloc(DD * 2);
  unsigned short* wb_q1   = (unsigned short*)alloc(DD * 2);
  unsigned short* wb_q2   = (unsigned short*)alloc(DD * 2);
  unsigned short* wb_qa   = (unsigned short*)alloc(DD * 2);
  unsigned short* wb_ka   = (unsigned short*)alloc(DD * 2);
  unsigned short* wb_va   = (unsigned short*)alloc(DD * 2);
  unsigned short* wb_oa   = (unsigned short*)alloc(DD * 2);
  unsigned short* wb_f1   = (unsigned short*)alloc(DD * 2 * 2);
  unsigned short* wb_f2   = (unsigned short*)alloc(DD * 2 * 2);

  const int TB = 256;
  dim3 blk(TB);
  dim3 grid_nd((unsigned)((ND_ + TB - 1) / TB));
  dim3 grid_gemm(N_ / 128, D_ / 16);            // 256 x 16
  dim3 grid_gemm_f1(N_ / 128, 2 * D_ / 16);     // 256 x 32

  // 0) weights -> bf16
  auto cvt = [&](const float* src, unsigned short* dst, size_t n) {
    k_cvt_bf16<<<dim3((unsigned)((n + TB - 1) / TB)), blk, 0, stream>>>(src, dst, n);
  };
  cvt(W_gcn, wb_gcn, DD);   cvt(W_gin1, wb_gin1, DD); cvt(W_gin2, wb_gin2, DD);
  cvt(Wq1, wb_q1, DD);      cvt(Wq2, wb_q2, DD);
  cvt(Wq_a, wb_qa, DD);     cvt(Wk_a, wb_ka, DD);     cvt(Wv_a, wb_va, DD);
  cvt(Wo_a, wb_oa, DD);     cvt(Wf1, wb_f1, DD * 2);  cvt(Wf2, wb_f2, DD * 2);

  // 1) degrees -> dinv (stored in deg)
  k_deg_init<<<dim3(N_ / TB), blk, 0, stream>>>(deg);
  k_deg_edges<<<dim3(E_ / TB), blk, 0, stream>>>(col, deg);
  k_dinv<<<dim3(N_ / TB), blk, 0, stream>>>(deg);

  // 2) GCN: xw = x @ W_gcn ; h_gcn = dinv^2*xw + scatter ; GIN agg in parallel
  k_gemm_bf16<<<grid_gemm, blk, 0, stream>>>(x, wb_gcn, nullptr, B1, N_, D_, D_, 0);
  k_gcn_init<<<grid_nd, blk, 0, stream>>>(B1, deg, B2, B3);
  k_scatter<<<dim3((unsigned)((size_t)E_ * 64 / TB)), blk, 0, stream>>>(row, col, deg, x, B1, B2, B3);

  // 3) e0 = BN(h_gcn)   (b_gcn cancels inside BN)
  k_bn_stats<<<dim3(D_), blk, 0, stream>>>(B2, mean, rstd, N_);
  k_bn_apply<<<grid_nd, blk, 0, stream>>>(B2, B2, mean, rstd, g_kv, be_kv, ND_);

  // 4) GIN: y = x + agg ; h1 = relu(y@W1+b1) ; h_gin = h1@W2 (+b2, cancels) ; e1 = BN
  k_add<<<grid_nd, blk, 0, stream>>>(x, B3, B3, ND_);
  k_gemm_bf16<<<grid_gemm, blk, 0, stream>>>(B3, wb_gin1, b_gin1, B1, N_, D_, D_, 1);
  k_gemm_bf16<<<grid_gemm, blk, 0, stream>>>(B1, wb_gin2, b_gin2, B3, N_, D_, D_, 0);
  k_bn_stats<<<dim3(D_), blk, 0, stream>>>(B3, mean, rstd, N_);
  k_bn_apply<<<grid_nd, blk, 0, stream>>>(B3, B3, mean, rstd, g_kv, be_kv, ND_);

  // 5) global expert normalization + kv mix
  k_gzero<<<dim3(1), blk, 0, stream>>>(gstats);
  k_reduce<<<dim3(4096), blk, 0, stream>>>(B2, ND_, gstats);
  k_reduce<<<dim3(4096), blk, 0, stream>>>(B3, ND_, gstats);
  k_gfinal<<<dim3(1), blk, 0, stream>>>(gstats, 2.0f * (float)ND_);
  k_kv<<<grid_nd, blk, 0, stream>>>(B2, B3, x, gstats, B4, ND_);

  // 6) q projection MLP
  k_gemm_bf16<<<grid_gemm, blk, 0, stream>>>(x, wb_q1, bq1, B1, N_, D_, D_, 1);
  k_gemm_bf16<<<grid_gemm, blk, 0, stream>>>(B1, wb_q2, bq2, B2, N_, D_, D_, 0);

  // 7) attention projections: Qa=B3, Ka=B1, Va=B2
  k_gemm_bf16<<<grid_gemm, blk, 0, stream>>>(B2, wb_qa, bq_a, B3, N_, D_, D_, 0);
  k_gemm_bf16<<<grid_gemm, blk, 0, stream>>>(B4, wb_ka, bk_a, B1, N_, D_, D_, 0);
  k_gemm_bf16<<<grid_gemm, blk, 0, stream>>>(B4, wb_va, bv_a, B2, N_, D_, D_, 0);

  // 8) flash attention -> O (B4)
  k_flash_attn<<<dim3(B_ * H_ * (S_ / 16) / 8), blk, 0, stream>>>(B3, B1, B2, B4);

  // 9) h_attn = O @ Wo_a + bo_a ; h = BN(x + h_attn)
  k_gemm_bf16<<<grid_gemm, blk, 0, stream>>>(B4, wb_oa, bo_a, B1, N_, D_, D_, 0);
  k_add<<<grid_nd, blk, 0, stream>>>(x, B1, B2, ND_);
  k_bn_stats<<<dim3(D_), blk, 0, stream>>>(B2, mean, rstd, N_);
  k_bn_apply<<<grid_nd, blk, 0, stream>>>(B2, B2, mean, rstd, g_at, be_at, ND_);

  // 10) FFN: ff1 = relu(h@Wf1+bf1) [Nx512] ; hf = ff1@Wf2+bf2 ; out = BN(h + hf)
  k_gemm_bf16<<<grid_gemm_f1, blk, 0, stream>>>(B2, wb_f1, bf1, B5, N_, D_, 2 * D_, 1);
  k_gemm_bf16<<<grid_gemm, blk, 0, stream>>>(B5, wb_f2, bf2, B3, N_, 2 * D_, D_, 0);
  k_add<<<grid_nd, blk, 0, stream>>>(B2, B3, B4, ND_);
  k_bn_stats<<<dim3(D_), blk, 0, stream>>>(B4, mean, rstd, N_);
  k_bn_apply<<<grid_nd, blk, 0, stream>>>(B4, (float*)d_out, mean, rstd, g2, be2, ND_);
}
